// IndividualAutoEncoder_14817637171531
// MI455X (gfx1250) — matrix operand order repaired and checked by reading, not compiled, not run on hardware
//
#include <hip/hip_runtime.h>
#include <hip/hip_bf16.h>

// ---------------------------------------------------------------------------
// Seq2seq RNN autoencoder for MI455X (gfx1250).
// B=64, S=512, V=4, H=1024.
// Persistent kernel: 32 WGs x 256 threads = 256 wave32s = one 16x16 output
// tile per wave for the M=64 x N=1024 recurrent GEMMs.
// bf16 WMMA (v_wmma_f32_16x16x32_bf16) with f32 accumulation; weights
// converted to bf16 once (14 MB, L2-resident).
// A-operand staged via GLOBAL_LOAD_ASYNC_TO_LDS_B128 (ASYNCcnt) with a
// double-buffered LDS ping-pong: one s_wait_asynccnt + one barrier per
// 128-wide K chunk. Grid-wide software barrier orders the 512-step recurrence.
// ---------------------------------------------------------------------------

typedef __attribute__((ext_vector_type(16))) __bf16 v16bf;
typedef __attribute__((ext_vector_type(8)))  __bf16 v8bf;
typedef __attribute__((ext_vector_type(8)))  float  v8f;

#define HB 1024
#define BB 64
#define SS 512
#define VV 4
#define NWG 32
#define KCH 128                    // K chunk (halves) staged per LDS buffer
#define LDS_STRIDE 136             // 128 + 8 halves pad: 68-bank row skew

// ---- workspace layout (bytes) ----
#define OFF_CNT   0ul                                   // [0]=cnt [1]=gen
#define OFF_W     256ul
#define W_ELEMS   (7ul * 1024ul * 1024ul)               // 7 HxH matrices
#define OFF_BIAS  (OFF_W + W_ELEMS * 2ul)               // c0,c1,d0,d1 f32[1024]
#define OFF_H0    (OFF_BIAS + 4ul * 1024ul * 4ul)
#define HBUF      (2ul * BB * HB * 2ul)                 // ping-pong bf16
#define OFF_H1    (OFF_H0 + HBUF)
#define OFF_HD0   (OFF_H1 + HBUF)
#define OFF_HD1   (OFF_HD0 + HBUF)
#define OFF_Y     (OFF_HD1 + HBUF)                      // bf16 64x1024
#define OFF_HD1F  (OFF_Y + (unsigned long)(BB * HB * 2)) // f32 64x1024

__device__ __forceinline__ unsigned short f2bf(float f) {
  unsigned u = __float_as_uint(f);
  unsigned r = (u + 0x7FFFu + ((u >> 16) & 1u)) >> 16;   // RNE
  return (unsigned short)r;
}

__device__ __forceinline__ float wred(float v) {
#pragma unroll
  for (int m = 16; m; m >>= 1) v += __shfl_xor(v, m, 32);
  return v;
}

__device__ __forceinline__ void gridbar(unsigned* cnt, volatile unsigned* gen) {
  __syncthreads();
  if (threadIdx.x == 0) {
    __threadfence();
    unsigned g = *gen;
    if (atomicAdd(cnt, 1u) == NWG - 1u) {
      *cnt = 0u;
      __threadfence();
      *gen = g + 1u;
    } else {
      while (*gen == g) { __builtin_amdgcn_s_sleep(1); }
    }
    __threadfence();
  }
  __syncthreads();
}

// Stage A[0:64][kk:kk+128] (bf16, row-major, row stride HB) into an LDS
// buffer (padded stride LDS_STRIDE) using async direct-to-LDS copies.
// 256 threads x 4 issues x 16B = 16 KB chunk. No VGPR data, tracked by
// ASYNCcnt.
__device__ __forceinline__ void async_copy_chunk(const unsigned short* __restrict__ A,
                                                 unsigned short* buf, int kk) {
  const int tid = threadIdx.x;
  const int row = tid >> 2;                         // 0..63
  const unsigned long long ab = (unsigned long long)A;
  const unsigned ldsrow = (unsigned)(unsigned long long)(buf + row * LDS_STRIDE);
  const unsigned grow   = (unsigned)(row * HB + kk) * 2u;
#pragma unroll
  for (int i = 0; i < 4; ++i) {
    const unsigned s   = (unsigned)(tid & 3) + 4u * (unsigned)i;  // segment 0..15
    const unsigned lds = ldsrow + s * 16u;
    const unsigned go  = grow + s * 16u;
    asm volatile("global_load_async_to_lds_b128 %0, %1, %2"
                 :: "v"(lds), "v"(go), "s"(ab) : "memory");
  }
}

// acc += A(64x1024 bf16, rows m0..m0+15) x W^T (W is [1024 out][1024 in] bf16
// row-major => already B-fragment-ready: two 8-K runs per lane, at 8h and 16+8h).
// Double-buffered LDS pipeline: prefetch chunk c+1 while computing chunk c.
__device__ __forceinline__ void gemm_tile(const unsigned short* __restrict__ A,
                                          const unsigned short* __restrict__ W,
                                          unsigned short* Asub,   // 2 buffers
                                          int m0, int n0, int lane, v8f& acc) {
  const int am    = m0 + (lane & 15);
  const int abase = (lane >> 4) << 3;  // lanes 16-31: K+8 (ISA A layout)
  const int n     = n0 + (lane & 15);
  const int kboff = (lane >> 4) << 3;  // lanes 16-31: K+8 (matches A layout)

  async_copy_chunk(A, Asub, 0);                       // prologue: chunk 0
  asm volatile("s_wait_asynccnt 0" ::: "memory");
  __syncthreads();

  for (int c = 0; c < HB / KCH; ++c) {
    unsigned short* cur = Asub + (c & 1) * (BB * LDS_STRIDE);
    if (c + 1 < HB / KCH)                             // async prefetch next
      async_copy_chunk(A, Asub + ((c + 1) & 1) * (BB * LDS_STRIDE),
                       (c + 1) * KCH);
#pragma unroll
    for (int kq = 0; kq < 4; ++kq) {
      const int kl = kq << 5;
      union { v16bf v; v8bf h[2]; } a;
      a.h[0] = *(const v8bf*)(cur + am * LDS_STRIDE + kl + abase);
      a.h[1] = *(const v8bf*)(cur + am * LDS_STRIDE + kl + abase + 16);
      union { v16bf v; v8bf h[2]; } bu;
      bu.h[0] = *(const v8bf*)(W + n * HB + c * KCH + kl + kboff);
      bu.h[1] = *(const v8bf*)(W + n * HB + c * KCH + kl + kboff + 16);
      const v16bf b = bu.v;
      acc = __builtin_amdgcn_wmma_f32_16x16x32_bf16(false, a.v, false, b,
                                                    (short)0, acc, false, false);
    }
    __builtin_prefetch(W + n * HB + c * KCH + KCH + kboff, 0, 3);
    asm volatile("s_wait_asynccnt 0" ::: "memory");   // chunk c+1 landed
    __syncthreads();                                  // all waves done with c
  }
}

// ---------------------------------------------------------------------------
__global__ void convert_weights_kernel(const float* w0, const float* w1,
                                       const float* w2, const float* w3,
                                       const float* w4, const float* w5,
                                       const float* w6, unsigned short* dst) {
  const float* srcs[7] = {w0, w1, w2, w3, w4, w5, w6};
  unsigned long i = (unsigned long)blockIdx.x * blockDim.x + threadIdx.x;
  if (i < W_ELEMS) dst[i] = f2bf(srcs[i >> 20][i & 1048575ul]);
}

__global__ void init_ws_kernel(unsigned char* ws,
                               const float* eb0a, const float* eb0b,
                               const float* eb1a, const float* eb1b,
                               const float* db0a, const float* db0b,
                               const float* db1a, const float* db1b) {
  unsigned i = blockIdx.x * blockDim.x + threadIdx.x;
  if (i == 0) { ((unsigned*)(ws + OFF_CNT))[0] = 0u; ((unsigned*)(ws + OFF_CNT))[1] = 0u; }
  float* bias = (float*)(ws + OFF_BIAS);
  if (i < 1024) {
    bias[i]          = eb0a[i] + eb0b[i];   // c0
    bias[1024 + i]   = eb1a[i] + eb1b[i];   // c1
    bias[2048 + i]   = db0a[i] + db0b[i];   // d0
    bias[3072 + i]   = db1a[i] + db1b[i];   // d1
  }
  unsigned short* hz = (unsigned short*)(ws + OFF_H0);  // h0,h1,hd0,hd1
  for (unsigned j = i; j < 4u * 2u * BB * HB; j += gridDim.x * blockDim.x) hz[j] = 0;
}

// ---------------------------------------------------------------------------
__global__ void __launch_bounds__(256, 1)
rnn_persistent_kernel(const float* __restrict__ x,      // (B,S,V)
                      const float* __restrict__ Wih0,   // enc_Wih0 (H,V) f32
                      const float* __restrict__ gamma,
                      const float* __restrict__ beta,
                      const float* __restrict__ projW,  // (V,H) f32
                      unsigned char* ws, float* __restrict__ out) {
  const int tid  = threadIdx.x;
  const int lane = tid & 31;
  const int wave = tid >> 5;
  const int gw   = blockIdx.x * 8 + wave;
  const int m0   = (wave & 3) * 16;
  const int n0   = (blockIdx.x * 2 + (wave >> 2)) * 16;

  __shared__ unsigned short Asub[2 * BB * LDS_STRIDE];   // double buffer, 34 KB

  unsigned* cnt = (unsigned*)(ws + OFF_CNT);
  volatile unsigned* gen = (volatile unsigned*)(ws + OFF_CNT + 4);
  const unsigned short* Wb  = (const unsigned short*)(ws + OFF_W);
  const unsigned short* Wenc_hh0 = Wb;
  const unsigned short* Wenc_ih1 = Wb + 1ul * 1048576;
  const unsigned short* Wenc_hh1 = Wb + 2ul * 1048576;
  const unsigned short* Wdec_ih0 = Wb + 3ul * 1048576;
  const unsigned short* Wdec_hh0 = Wb + 4ul * 1048576;
  const unsigned short* Wdec_ih1 = Wb + 5ul * 1048576;
  const unsigned short* Wdec_hh1 = Wb + 6ul * 1048576;
  const float* bias = (const float*)(ws + OFF_BIAS);
  unsigned short* h0  = (unsigned short*)(ws + OFF_H0);
  unsigned short* h1  = (unsigned short*)(ws + OFF_H1);
  unsigned short* hd0 = (unsigned short*)(ws + OFF_HD0);
  unsigned short* hd1 = (unsigned short*)(ws + OFF_HD1);
  unsigned short* yb  = (unsigned short*)(ws + OFF_Y);
  float* hd1f = (float*)(ws + OFF_HD1F);

  // per-lane hoists (n is fixed for this lane across the whole kernel)
  const int nme = n0 + (lane & 15);
  const float w00 = Wih0[nme * 4 + 0], w01 = Wih0[nme * 4 + 1];
  const float w02 = Wih0[nme * 4 + 2], w03 = Wih0[nme * 4 + 3];
  const float c0n = bias[nme], c1n = bias[1024 + nme];
  const float d0n = bias[2048 + nme], d1n = bias[3072 + nme];
  const int mhi = (lane >> 4) << 3;

  // ================= encoder: fused layer0 + layer1 =================
  for (int t = 0; t < SS; ++t) {
    const int wp = t & 1, rp = wp ^ 1;
    // ---- layer 0: tanh(x[t] @ Wih0^T + h0 @ Whh0^T + c0) ----
    v8f acc;
#pragma unroll
    for (int i = 0; i < 8; ++i) acc[i] = 0.f;
    gemm_tile(h0 + rp * (BB * HB), Wenc_hh0, Asub, m0, n0, lane, acc);
#pragma unroll
    for (int r = 0; r < 8; ++r) {
      const int m = m0 + r + mhi;
      const float* xp = x + ((unsigned long)m * SS + t) * VV;
      float v = acc[r] + c0n + xp[0] * w00 + xp[1] * w01 + xp[2] * w02 + xp[3] * w03;
      h0[wp * (BB * HB) + m * HB + nme] = f2bf(tanhf(v));
    }
    gridbar(cnt, gen);
    // ---- layer 1: tanh(h0_new @ Wih1^T + h1 @ Whh1^T + c1) ----
#pragma unroll
    for (int i = 0; i < 8; ++i) acc[i] = 0.f;
    gemm_tile(h0 + wp * (BB * HB), Wenc_ih1, Asub, m0, n0, lane, acc);
    gemm_tile(h1 + rp * (BB * HB), Wenc_hh1, Asub, m0, n0, lane, acc);
#pragma unroll
    for (int r = 0; r < 8; ++r) {
      const int m = m0 + r + mhi;
      h1[wp * (BB * HB) + m * HB + nme] = f2bf(tanhf(acc[r] + c1n));
    }
    gridbar(cnt, gen);
  }

  const unsigned short* h1fin = h1 + ((SS - 1) & 1) * (BB * HB);  // y1[-1]

  // ================= decoder: L0 + L1 + LN + proj + log_softmax =============
  for (int t = 0; t < SS; ++t) {
    const int wp = t & 1, rp = wp ^ 1;
    const unsigned short* xd = (t == 0) ? h1fin : yb;
    // ---- layer 0 ----
    v8f acc;
#pragma unroll
    for (int i = 0; i < 8; ++i) acc[i] = 0.f;
    gemm_tile(xd, Wdec_ih0, Asub, m0, n0, lane, acc);
    gemm_tile(hd0 + rp * (BB * HB), Wdec_hh0, Asub, m0, n0, lane, acc);
#pragma unroll
    for (int r = 0; r < 8; ++r) {
      const int m = m0 + r + mhi;
      hd0[wp * (BB * HB) + m * HB + nme] = f2bf(tanhf(acc[r] + d0n));
    }
    gridbar(cnt, gen);
    // ---- layer 1 (store bf16 for recurrence + f32 for LN) ----
#pragma unroll
    for (int i = 0; i < 8; ++i) acc[i] = 0.f;
    gemm_tile(hd0 + wp * (BB * HB), Wdec_ih1, Asub, m0, n0, lane, acc);
    gemm_tile(hd1 + rp * (BB * HB), Wdec_hh1, Asub, m0, n0, lane, acc);
#pragma unroll
    for (int r = 0; r < 8; ++r) {
      const int m = m0 + r + mhi;
      const float v = tanhf(acc[r] + d1n);
      hd1[wp * (BB * HB) + m * HB + nme] = f2bf(v);
      hd1f[m * HB + nme] = v;
    }
    gridbar(cnt, gen);
    // ---- LayerNorm + projection (V=4) + log_softmax: one wave per row ----
    if (gw < BB) {
      const int b = gw;
      const float* hrow = hd1f + b * HB;
      float s = 0.f, s2 = 0.f;
#pragma unroll 4
      for (int j = 0; j < 32; ++j) {
        const float v = hrow[lane + 32 * j];
        s += v; s2 += v * v;
      }
      s = wred(s); s2 = wred(s2);
      const float mu  = s * (1.f / HB);
      const float var = s2 * (1.f / HB) - mu * mu;
      const float inv = rsqrtf(var + 1e-5f);
      float l0 = 0.f, l1 = 0.f, l2 = 0.f, l3 = 0.f;
#pragma unroll 4
      for (int j = 0; j < 32; ++j) {
        const int n = lane + 32 * j;
        const float yv = (hrow[n] - mu) * inv * gamma[n] + beta[n];
        yb[b * HB + n] = f2bf(yv);                 // next-step decoder input
        l0 += yv * projW[n];
        l1 += yv * projW[HB + n];
        l2 += yv * projW[2 * HB + n];
        l3 += yv * projW[3 * HB + n];
      }
      l0 = wred(l0); l1 = wred(l1); l2 = wred(l2); l3 = wred(l3);
      if (lane == 0) {
        const float mx = fmaxf(fmaxf(l0, l1), fmaxf(l2, l3));
        const float lse = logf(expf(l0 - mx) + expf(l1 - mx) +
                               expf(l2 - mx) + expf(l3 - mx)) + mx;
        float* op = out + (((unsigned long)(BB - 1 - b)) * SS + t) * VV;  // out[::-1]
        op[0] = l0 - lse;
        op[1] = l1 - lse;
        op[2] = l2 - lse;
        op[3] = l3 - lse;
      }
    }
    gridbar(cnt, gen);
  }
}

// ---------------------------------------------------------------------------
extern "C" void kernel_launch(void* const* d_in, const int* in_sizes, int n_in,
                              void* d_out, int out_size, void* d_ws, size_t ws_size,
                              hipStream_t stream) {
  const float* x        = (const float*)d_in[0];
  const float* eWih0    = (const float*)d_in[1];
  const float* eWhh0    = (const float*)d_in[2];
  const float* ebih0    = (const float*)d_in[3];
  const float* ebhh0    = (const float*)d_in[4];
  const float* eWih1    = (const float*)d_in[5];
  const float* eWhh1    = (const float*)d_in[6];
  const float* ebih1    = (const float*)d_in[7];
  const float* ebhh1    = (const float*)d_in[8];
  const float* dWih0    = (const float*)d_in[9];
  const float* dWhh0    = (const float*)d_in[10];
  const float* dbih0    = (const float*)d_in[11];
  const float* dbhh0    = (const float*)d_in[12];
  const float* dWih1    = (const float*)d_in[13];
  const float* dWhh1    = (const float*)d_in[14];
  const float* dbih1    = (const float*)d_in[15];
  const float* dbhh1    = (const float*)d_in[16];
  const float* ln_gamma = (const float*)d_in[17];
  const float* ln_beta  = (const float*)d_in[18];
  const float* projW    = (const float*)d_in[19];
  unsigned char* ws = (unsigned char*)d_ws;
  float* out = (float*)d_out;

  // 1) f32 -> bf16 weight conversion (14 MB, lives in L2 for the whole run)
  {
    const unsigned long n = W_ELEMS;
    const int blocks = (int)((n + 255) / 256);
    convert_weights_kernel<<<blocks, 256, 0, stream>>>(
        eWhh0, eWih1, eWhh1, dWih0, dWhh0, dWih1, dWhh1,
        (unsigned short*)(ws + OFF_W));
  }
  // 2) combined biases + zero hidden states + barrier counters
  init_ws_kernel<<<512, 256, 0, stream>>>(ws, ebih0, ebhh0, ebih1, ebhh1,
                                          dbih0, dbhh0, dbih1, dbhh1);
  // 3) persistent fused seq2seq RNN
  rnn_persistent_kernel<<<NWG, 256, 0, stream>>>(x, eWih0, ln_gamma, ln_beta,
                                                 projW, ws, out);
}
